// HeteroGAT_1460288881211
// MI455X (gfx1250) — matrix operand order repaired and checked by reading, not compiled, not run on hardware
//
#include <hip/hip_runtime.h>

#define NP 50000
#define NS 2000
#define NC 500
#define DD 128
#define EGAT 500000
#define EGCN 1000000
#define NEG_SLOPE 0.2f

typedef __attribute__((ext_vector_type(16))) __bf16 v16bf;
typedef __attribute__((ext_vector_type(8)))  float  v8f;
typedef __attribute__((ext_vector_type(4)))  unsigned v4u;

static __device__ __forceinline__ unsigned short f32_to_bf16(float f) {
  union { float f; unsigned u; } x; x.f = f;
  unsigned r = x.u + 0x7FFFu + ((x.u >> 16) & 1u);
  return (unsigned short)(r >> 16);
}

// order-preserving float<->uint mapping for atomic max
static __device__ __forceinline__ unsigned f2ord(float f) {
  unsigned b = __float_as_uint(f);
  return (b & 0x80000000u) ? ~b : (b | 0x80000000u);
}
static __device__ __forceinline__ float ord2f(unsigned k) {
  unsigned b = (k & 0x80000000u) ? (k ^ 0x80000000u) : ~k;
  return __uint_as_float(b);
}

// ---------------------------------------------------------------------------
// GEMM: C[M,128] = A[M,K] @ W[K,128] (+bias). bf16 WMMA, f32 accumulate.
// 128 threads = 4 waves; each wave computes a 16x128 strip; block does 64 rows.
// W and A are staged into LDS pre-swizzled into WMMA fragment order, so each
// fragment load is two ds_load_b128 per lane.
// ---------------------------------------------------------------------------
#define KBLK_MAX 5  // ceil(160/32): K in {9,96,128,138}
__global__ __launch_bounds__(128) void gemm_bias_wmma(
    const float* __restrict__ A, const float* __restrict__ W,
    const float* __restrict__ bias, float* __restrict__ C, int M, int K) {
  // B fragments: [kblk][tile(8)][lane(32)][v(8)] dwords (bf16 pairs)
  __shared__ unsigned sWp[KBLK_MAX * 8 * 32 * 8];  // 40 KB
  // A fragments: [wave(4)][lane(32)][v(8)] dwords
  __shared__ unsigned sAp[4 * 32 * 8];             // 4 KB
  const int Kpad = (K + 31) & ~31;
  const int nkb  = Kpad >> 5;
  const int tid  = threadIdx.x;
  const int wave = tid >> 5, lane = tid & 31;
  const int mlane = lane & 15, khalf = lane >> 4;
  const int row0 = blockIdx.x * 64;

  // pack W into B-fragment order: lane l (col group), VGPR v holds
  // K = kb*32 + (l>>4)*16 + 2v, 2v+1 at column n = t*16 + (l&15)
  for (int i = tid; i < nkb * 2048; i += 128) {
    int v  = i & 7;
    int l  = (i >> 3) & 31;
    int t  = (i >> 8) & 7;
    int kb = i >> 11;
    int k  = kb * 32 + (l >> 4) * 16 + 2 * v;
    int n  = t * 16 + (l & 15);
    unsigned lo = (k < K)     ? (unsigned)f32_to_bf16(W[k * 128 + n])       : 0u;
    unsigned hi = (k + 1 < K) ? (unsigned)f32_to_bf16(W[(k + 1) * 128 + n]) : 0u;
    sWp[i] = lo | (hi << 16);
  }
  __syncthreads();

  v8f acc[8] = {};

  for (int kb = 0; kb < nkb; ++kb) {
    __syncthreads();
    // pack A tile into A-fragment order: wave w, lane l -> row w*16+(l&15),
    // VGPR v holds K = kb*32 + (l>>4)*16 + 2v, +1
    for (int i = tid; i < 1024; i += 128) {
      int v  = i & 7;
      int l  = (i >> 3) & 31;
      int wv = i >> 8;
      int r  = row0 + wv * 16 + (l & 15);
      int kk = kb * 32 + (l >> 4) * 16 + 2 * v;
      unsigned lo = (r < M && kk < K)     ? (unsigned)f32_to_bf16(A[(size_t)r * K + kk])     : 0u;
      unsigned hi = (r < M && kk + 1 < K) ? (unsigned)f32_to_bf16(A[(size_t)r * K + kk + 1]) : 0u;
      sAp[i] = lo | (hi << 16);
    }
    __syncthreads();

    union { v16bf v; v4u q[2]; } af;
    const unsigned* ap = &sAp[(wave * 32 + lane) * 8];
    af.q[0] = *(const v4u*)ap;
    af.q[1] = *(const v4u*)(ap + 4);
#pragma unroll
    for (int t = 0; t < 8; ++t) {
      union { v16bf v; v4u q[2]; } bfr;
      const unsigned* bp = &sWp[((kb * 8 + t) * 32 + lane) * 8];
      bfr.q[0] = *(const v4u*)bp;
      bfr.q[1] = *(const v4u*)(bp + 4);
      acc[t] = __builtin_amdgcn_wmma_f32_16x16x32_bf16(
          false, af.v, false, bfr.v, (short)0, acc[t], false, false);
    }
  }

  // C/D layout: VGPR i, lanes0-15 -> M=i, lanes16-31 -> M=8+i; N = lane&15
  const int mbase = row0 + wave * 16 + khalf * 8;
#pragma unroll
  for (int t = 0; t < 8; ++t) {
    const int n = t * 16 + mlane;
    const float bv = bias ? bias[n] : 0.f;
#pragma unroll
    for (int i = 0; i < 8; ++i) {
      int r = mbase + i;
      if (r < M) C[r * 128 + n] = acc[t][i] + bv;
    }
  }
}

// ---------------------------------------------------------------------------
__global__ void fill_f32_kernel(float* p, float v, int n) {
  int i = blockIdx.x * blockDim.x + threadIdx.x;
  if (i < n) p[i] = v;
}
__global__ void fill_u32_kernel(unsigned* p, unsigned v, int n) {
  int i = blockIdx.x * blockDim.x + threadIdx.x;
  if (i < n) p[i] = v;
}
__global__ void rows_bias3_kernel(float* out, const float* b0, const float* b1,
                                  const float* b2, int total) {
  int i = blockIdx.x * blockDim.x + threadIdx.x;
  if (i >= total) return;
  int c = i & 127;
  float v = b0[c];
  if (b1) v += b1[c];
  if (b2) v += b2[c];
  out[i] = v;
}
__global__ void relu_kernel(float* x, int n) {
  int i = blockIdx.x * blockDim.x + threadIdx.x;
  if (i < n) x[i] = fmaxf(x[i], 0.f);
}

// a[n,h] = sum_c H[n, h*32+c] * att[h,c]
__global__ void att_scores_kernel(const float* __restrict__ H,
                                  const float* __restrict__ att,
                                  float* __restrict__ out, int N) {
  int i = blockIdx.x * blockDim.x + threadIdx.x;
  if (i >= N * 4) return;
  int n = i >> 2, h = i & 3;
  const float* hp = H + n * 128 + h * 32;
  const float* ap = att + h * 32;
  float s = 0.f;
#pragma unroll
  for (int c = 0; c < 32; ++c) s += hp[c] * ap[c];
  out[i] = s;
}

static __device__ __forceinline__ float lrelu(float v) {
  return v > 0.f ? v : NEG_SLOPE * v;
}

__global__ void gat_edge_max_kernel(const int* __restrict__ ei, int E,
                                    const float* __restrict__ as_,
                                    const float* __restrict__ ad_,
                                    unsigned* __restrict__ mu) {
  int e = blockIdx.x * blockDim.x + threadIdx.x;
  if (e >= E) return;
  int s = ei[e], d = ei[E + e];
#pragma unroll
  for (int h = 0; h < 4; ++h) {
    float v = lrelu(as_[s * 4 + h] + ad_[d * 4 + h]);
    atomicMax(&mu[d * 4 + h], f2ord(v));
  }
}

__global__ void gat_edge_exp_kernel(const int* __restrict__ ei, int E,
                                    const float* __restrict__ as_,
                                    const float* __restrict__ ad_,
                                    const unsigned* __restrict__ mu,
                                    float* __restrict__ ex,
                                    float* __restrict__ den) {
  int e = blockIdx.x * blockDim.x + threadIdx.x;
  if (e >= E) return;
  int s = ei[e], d = ei[E + e];
#pragma unroll
  for (int h = 0; h < 4; ++h) {
    float v = lrelu(as_[s * 4 + h] + ad_[d * 4 + h]);
    float x = __expf(v - ord2f(mu[d * 4 + h]));
    ex[e * 4 + h] = x;
    atomicAdd(&den[d * 4 + h], x);
  }
}

// one wave32 per edge: lane = channel within head, loop heads
__global__ __launch_bounds__(256) void gat_edge_scatter_kernel(
    const int* __restrict__ ei, int E, const float* __restrict__ ex,
    const float* __restrict__ den, const float* __restrict__ Hs,
    float* __restrict__ out) {
  int e = blockIdx.x * 8 + (threadIdx.x >> 5);
  int lane = threadIdx.x & 31;
  if (e >= E) return;
  int s = ei[e], d = ei[E + e];
#pragma unroll
  for (int h = 0; h < 4; ++h) {
    float alpha = ex[e * 4 + h] / (den[d * 4 + h] + 1e-16f);
    int c = h * 32 + lane;
    atomicAdd(&out[d * 128 + c], Hs[s * 128 + c] * alpha);
  }
}

__global__ void gcn_deg_kernel(const int* __restrict__ ei, float* __restrict__ deg, int E) {
  int e = blockIdx.x * blockDim.x + threadIdx.x;
  if (e >= E) return;
  atomicAdd(&deg[ei[E + e]], 1.f);
}

__global__ __launch_bounds__(256) void gcn_scatter_kernel(
    const int* __restrict__ ei, int E, const float* __restrict__ H,
    const float* __restrict__ deg, float* __restrict__ out) {
  int e = blockIdx.x * 8 + (threadIdx.x >> 5);
  int lane = threadIdx.x & 31;
  if (e >= E) return;
  int s = ei[e], d = ei[E + e];
  float norm = rsqrtf(deg[s]) * rsqrtf(deg[d]);
#pragma unroll
  for (int i = 0; i < 4; ++i) {
    int c = i * 32 + lane;
    atomicAdd(&out[d * 128 + c], H[s * 128 + c] * norm);
  }
}

__global__ void gcn_self_kernel(const float* __restrict__ H,
                                const float* __restrict__ deg,
                                float* __restrict__ out, int total) {
  int i = blockIdx.x * blockDim.x + threadIdx.x;
  if (i >= total) return;
  out[i] += H[i] / deg[i >> 7];  // dinv^2 = 1/deg
}

// ---------------------------------------------------------------------------
static inline int cdiv(int a, int b) { return (a + b - 1) / b; }

struct Ws {
  float *cur_xp, *cur_xs, *cur_xc, *new_xp, *new_xs, *new_xc;
  float *Hsrc, *Hdst, *a_src, *a_dst, *den, *ex, *deg;
  unsigned* mu;
};

static void run_gat(hipStream_t st, Ws& w, const float* x_src, int Ns,
                    const float* x_dst, int Nd, const int* ei, int E,
                    const float* Wd, const float* Wsrc, const float* att_d,
                    const float* att_s, float* out_accum) {
  gemm_bias_wmma<<<cdiv(Ns, 64), 128, 0, st>>>(x_src, Wsrc, nullptr, w.Hsrc, Ns, 128);
  gemm_bias_wmma<<<cdiv(Nd, 64), 128, 0, st>>>(x_dst, Wd, nullptr, w.Hdst, Nd, 128);
  att_scores_kernel<<<cdiv(Ns * 4, 256), 256, 0, st>>>(w.Hsrc, att_s, w.a_src, Ns);
  att_scores_kernel<<<cdiv(Nd * 4, 256), 256, 0, st>>>(w.Hdst, att_d, w.a_dst, Nd);
  fill_u32_kernel<<<cdiv(Nd * 4, 256), 256, 0, st>>>(w.mu, 0u, Nd * 4);
  fill_f32_kernel<<<cdiv(Nd * 4, 256), 256, 0, st>>>(w.den, 0.f, Nd * 4);
  gat_edge_max_kernel<<<cdiv(E, 256), 256, 0, st>>>(ei, E, w.a_src, w.a_dst, w.mu);
  gat_edge_exp_kernel<<<cdiv(E, 256), 256, 0, st>>>(ei, E, w.a_src, w.a_dst, w.mu, w.ex, w.den);
  gat_edge_scatter_kernel<<<cdiv(E, 8), 256, 0, st>>>(ei, E, w.ex, w.den, w.Hsrc, out_accum);
}

extern "C" void kernel_launch(void* const* d_in, const int* in_sizes, int n_in,
                              void* d_out, int out_size, void* d_ws, size_t ws_size,
                              hipStream_t stream) {
  const float* xp_in = (const float*)d_in[0];
  const float* xs_in = (const float*)d_in[1];
  const float* xc_in = (const float*)d_in[2];
  const int* ei_follows = (const int*)d_in[3];
  const int* ei_pc = (const int*)d_in[4];
  const int* ei_cp = (const int*)d_in[5];
  const int* ei_ps = (const int*)d_in[6];
  const int* ei_sp = (const int*)d_in[7];
  // params flattened in JAX pytree order (sorted dict keys at every level)
  void* const* p = d_in + 8;
  auto F = [&](int i) { return (const float*)p[i]; };
  // in: condition{W,b}, patient{W,b}, signature{W,b}
  const float *inW_c = F(0), *inb_c = F(1), *inW_p = F(2), *inb_p = F(3),
              *inW_s = F(4), *inb_s = F(5);
  // layer base 6 + 22*L : cp{Wd,Ws,att_d,att_s,b}, follows{W,b},
  //                       pc{...}, ps{...}, sp{...}
  const int LBASE = 6, LSTRIDE = 22;
  // out: condition{W,b}, patient{W,b}, signature{W,b}
  const int OB = LBASE + 3 * LSTRIDE;
  const float *oW_c = F(OB + 0), *ob_c = F(OB + 1), *oW_p = F(OB + 2),
              *ob_p = F(OB + 3), *oW_s = F(OB + 4), *ob_s = F(OB + 5);

  // workspace carve-up
  size_t off = 0;
  auto alloc = [&](size_t nfloats) {
    char* r = (char*)d_ws + off;
    off += (nfloats * 4 + 255) & ~(size_t)255;
    return (float*)r;
  };
  Ws w;
  w.cur_xp = alloc((size_t)NP * 128); w.cur_xs = alloc((size_t)NS * 128);
  w.cur_xc = alloc((size_t)NC * 128);
  w.new_xp = alloc((size_t)NP * 128); w.new_xs = alloc((size_t)NS * 128);
  w.new_xc = alloc((size_t)NC * 128);
  w.Hsrc = alloc((size_t)NP * 128);  w.Hdst = alloc((size_t)NP * 128);
  w.a_src = alloc((size_t)NP * 4);   w.a_dst = alloc((size_t)NP * 4);
  w.mu = (unsigned*)alloc((size_t)NP * 4);
  w.den = alloc((size_t)NP * 4);
  w.ex = alloc((size_t)EGAT * 4);
  w.deg = alloc((size_t)NP);

  hipStream_t st = stream;
  // input projections
  gemm_bias_wmma<<<cdiv(NP, 64), 128, 0, st>>>(xp_in, inW_p, inb_p, w.cur_xp, NP, 138);
  gemm_bias_wmma<<<cdiv(NS, 64), 128, 0, st>>>(xs_in, inW_s, inb_s, w.cur_xs, NS, 96);
  gemm_bias_wmma<<<cdiv(NC, 64), 128, 0, st>>>(xc_in, inW_c, inb_c, w.cur_xc, NC, 9);

  for (int L = 0; L < 3; ++L) {
    const int B = LBASE + L * LSTRIDE;
    const float *cpWd = F(B + 0), *cpWs = F(B + 1), *cpAd = F(B + 2),
                *cpAs = F(B + 3), *cpb = F(B + 4);
    const float *folW = F(B + 5), *folb = F(B + 6);
    const float *pcWd = F(B + 7), *pcWs = F(B + 8), *pcAd = F(B + 9),
                *pcAs = F(B + 10), *pcb = F(B + 11);
    const float *psWd = F(B + 12), *psWs = F(B + 13), *psAd = F(B + 14),
                *psAs = F(B + 15), *psb = F(B + 16);
    const float *spWd = F(B + 17), *spWs = F(B + 18), *spAd = F(B + 19),
                *spAs = F(B + 20), *spb = F(B + 21);

    // init accumulators with summed biases
    rows_bias3_kernel<<<cdiv(NP * 128, 256), 256, 0, st>>>(w.new_xp, folb, cpb, spb, NP * 128);
    rows_bias3_kernel<<<cdiv(NC * 128, 256), 256, 0, st>>>(w.new_xc, pcb, nullptr, nullptr, NC * 128);
    rows_bias3_kernel<<<cdiv(NS * 128, 256), 256, 0, st>>>(w.new_xs, psb, nullptr, nullptr, NS * 128);

    // GCN (patient->patient)
    gemm_bias_wmma<<<cdiv(NP, 64), 128, 0, st>>>(w.cur_xp, folW, nullptr, w.Hsrc, NP, 128);
    fill_f32_kernel<<<cdiv(NP, 256), 256, 0, st>>>(w.deg, 1.f, NP);
    gcn_deg_kernel<<<cdiv(EGCN, 256), 256, 0, st>>>(ei_follows, w.deg, EGCN);
    gcn_scatter_kernel<<<cdiv(EGCN, 8), 256, 0, st>>>(ei_follows, EGCN, w.Hsrc, w.deg, w.new_xp);
    gcn_self_kernel<<<cdiv(NP * 128, 256), 256, 0, st>>>(w.Hsrc, w.deg, w.new_xp, NP * 128);

    // GATs into patient
    run_gat(st, w, w.cur_xc, NC, w.cur_xp, NP, ei_cp, EGAT, cpWd, cpWs, cpAd, cpAs, w.new_xp);
    run_gat(st, w, w.cur_xs, NS, w.cur_xp, NP, ei_sp, EGAT, spWd, spWs, spAd, spAs, w.new_xp);
    // GATs into condition / signature
    run_gat(st, w, w.cur_xp, NP, w.cur_xc, NC, ei_pc, EGAT, pcWd, pcWs, pcAd, pcAs, w.new_xc);
    run_gat(st, w, w.cur_xp, NP, w.cur_xs, NS, ei_ps, EGAT, psWd, psWs, psAd, psAs, w.new_xs);

    relu_kernel<<<cdiv(NP * 128, 256), 256, 0, st>>>(w.new_xp, NP * 128);
    relu_kernel<<<cdiv(NC * 128, 256), 256, 0, st>>>(w.new_xc, NC * 128);
    relu_kernel<<<cdiv(NS * 128, 256), 256, 0, st>>>(w.new_xs, NS * 128);

    float* t;
    t = w.cur_xp; w.cur_xp = w.new_xp; w.new_xp = t;
    t = w.cur_xs; w.cur_xs = w.new_xs; w.new_xs = t;
    t = w.cur_xc; w.cur_xc = w.new_xc; w.new_xc = t;
  }

  // output projections; d_out = [condition | patient | signature] (pytree order)
  float* out = (float*)d_out;
  gemm_bias_wmma<<<cdiv(NC, 64), 128, 0, st>>>(w.cur_xc, oW_c, ob_c, out, NC, 128);
  gemm_bias_wmma<<<cdiv(NP, 64), 128, 0, st>>>(w.cur_xp, oW_p, ob_p, out + (size_t)NC * 128, NP, 128);
  gemm_bias_wmma<<<cdiv(NS, 64), 128, 0, st>>>(w.cur_xs, oW_s, ob_s, out + (size_t)(NC + NP) * 128, NS, 128);
}
